// WMTransformerLayer_23622320128491
// MI455X (gfx1250) — matrix operand order repaired and checked by reading, not compiled, not run on hardware
//
#include <hip/hip_runtime.h>

// ---------------------------------------------------------------------------
// Transformer layer for MI455X (gfx1250, wave32, WMMA).
// GEMMs + attention einsums on v_wmma_f32_16x16x32_f16 (f32 accum).
// LDS layouts give contiguous ds_load_b128 fragment loads; A/K tiles use
// gfx1250 async global->LDS loads (ASYNCcnt). Softmax reductions use DPP.
// ---------------------------------------------------------------------------

typedef _Float16 half_t;
typedef __attribute__((ext_vector_type(16))) _Float16 v16h;
typedef __attribute__((ext_vector_type(8)))  _Float16 v8h;
typedef __attribute__((ext_vector_type(8)))  float    v8f;
typedef __attribute__((ext_vector_type(4)))  int      v4i;

constexpr int Bc = 2;
constexpr int T  = 2048;
constexpr int D  = 1024;
constexpr int H  = 16;
constexpr int DH = 64;
constexpr int NB = DH / 2;   // 32 rotary pairs

#if defined(__has_builtin)
#if __has_builtin(__builtin_amdgcn_global_load_async_to_lds_b128) && \
    __has_builtin(__builtin_amdgcn_s_wait_asynccnt)
#define USE_ASYNC_LDS 1
#endif
#endif

static __device__ inline void copy_b128_g2l(const half_t* __restrict__ g,
                                            half_t* __restrict__ l) {
#if defined(USE_ASYNC_LDS)
    __builtin_amdgcn_global_load_async_to_lds_b128(
        (__attribute__((address_space(1))) v4i*)(half_t*)g,
        (__attribute__((address_space(3))) v4i*)l, 0, 0);
#else
    *(v8h*)l = *(const v8h*)g;
#endif
}

static __device__ inline void wait_g2l() {
#if defined(USE_ASYNC_LDS)
    __builtin_amdgcn_s_wait_asynccnt(0);
#endif
}

// ---- VALU (DPP) butterfly reductions within each 16-lane half-row ----
template<int CTRL>
static __device__ inline float dpp_perm(float x) {
    return __builtin_bit_cast(float,
        __builtin_amdgcn_mov_dpp(__builtin_bit_cast(int, x), CTRL, 0xF, 0xF, true));
}
static __device__ inline float red_max16(float v) {
    v = fmaxf(v, dpp_perm<0xB1>(v));    // quad_perm(1,0,3,2)  : xor 1
    v = fmaxf(v, dpp_perm<0x4E>(v));    // quad_perm(2,3,0,1)  : xor 2
    v = fmaxf(v, dpp_perm<0x141>(v));   // row_half_mirror     : pairs 4-groups
    v = fmaxf(v, dpp_perm<0x140>(v));   // row_mirror          : pairs 8-groups
    return v;
}
static __device__ inline float red_sum16(float v) {
    v += dpp_perm<0xB1>(v);
    v += dpp_perm<0x4E>(v);
    v += dpp_perm<0x141>(v);
    v += dpp_perm<0x140>(v);
    return v;
}

static __device__ inline v8f v8f_zero() {
    v8f v;
    #pragma unroll
    for (int i = 0; i < 8; i++) v[i] = 0.0f;
    return v;
}

// 16 contiguous halves -> one v16h fragment (2x ds_load_b128)
static __device__ inline v16h load_frag16(const half_t* __restrict__ p) {
    v8h lo = *(const v8h*)p;
    v8h hi = *(const v8h*)(p + 8);
    return __builtin_shufflevector(lo, hi, 0, 1, 2, 3, 4, 5, 6, 7,
                                           8, 9, 10, 11, 12, 13, 14, 15);
}

// two contiguous 8-half chunks -> one v16h fragment (A-matrix layout)
static __device__ inline v16h load_frag16_split(const half_t* __restrict__ p0,
                                                const half_t* __restrict__ p1) {
    v8h lo = *(const v8h*)p0;
    v8h hi = *(const v8h*)p1;
    return __builtin_shufflevector(lo, hi, 0, 1, 2, 3, 4, 5, 6, 7,
                                           8, 9, 10, 11, 12, 13, 14, 15);
}

// ---------------------------------------------------------------------------
// Elementwise f32 -> f16 convert (weights, V)
// ---------------------------------------------------------------------------
__global__ void f32_to_f16_kernel(const float* __restrict__ in,
                                  half_t* __restrict__ out, int n) {
    int i = blockIdx.x * blockDim.x + threadIdx.x;
    if (i < n) out[i] = (half_t)in[i];
}

// ---------------------------------------------------------------------------
// LayerNorm: one block per row of D=1024, fp32 math, f16 output.
// ---------------------------------------------------------------------------
__global__ void __launch_bounds__(256)
layernorm_kernel(const float* __restrict__ x, const float* __restrict__ g,
                 const float* __restrict__ bb, half_t* __restrict__ out) {
    __shared__ float red1[256];
    __shared__ float red2[256];
    const int row = blockIdx.x;
    const int tid = threadIdx.x;
    const float* xr = x + (size_t)row * D;
    float s = 0.0f, s2 = 0.0f;
    for (int i = tid; i < D; i += 256) { float v = xr[i]; s += v; s2 += v * v; }
    red1[tid] = s; red2[tid] = s2;
    __syncthreads();
    for (int off = 128; off > 0; off >>= 1) {
        if (tid < off) { red1[tid] += red1[tid + off]; red2[tid] += red2[tid + off]; }
        __syncthreads();
    }
    const float mean = red1[0] * (1.0f / D);
    const float var  = red2[0] * (1.0f / D) - mean * mean;
    const float inv  = rsqrtf(var + 1e-5f);
    half_t* orow = out + (size_t)row * D;
    for (int i = tid; i < D; i += 256)
        orow[i] = (half_t)((xr[i] - mean) * inv * g[i] + bb[i]);
}

// ---------------------------------------------------------------------------
// RoPE on Q,K (interleaved pairs); Q gets pre-scaled by 1/sqrt(DH). f16 out.
// ---------------------------------------------------------------------------
__global__ void rope_qk_kernel(const float* __restrict__ Qf, const float* __restrict__ Kf,
                               const float* __restrict__ cosb, const float* __restrict__ sinb,
                               half_t* __restrict__ Q16, half_t* __restrict__ K16, int total) {
    int idx = blockIdx.x * blockDim.x + threadIdx.x;
    if (idx >= total) return;
    const int nb = idx % NB;
    const int hh = (idx / NB) % H;
    const int t  = (idx / (NB * H)) % T;
    const int bb = idx / (NB * H * T);
    const float c = cosb[(size_t)t * NB + nb];   // broadcast plane (b=0,h=0)
    const float s = sinb[(size_t)t * NB + nb];
    const size_t base = ((size_t)(bb * T + t)) * D + hh * DH + 2 * nb;
    const float q1 = Qf[base], q2 = Qf[base + 1];
    Q16[base]     = (half_t)((q1 * c - q2 * s) * 0.125f);   // 1/sqrt(64)
    Q16[base + 1] = (half_t)((q1 * s + q2 * c) * 0.125f);
    const float k1 = Kf[base], k2 = Kf[base + 1];
    K16[base]     = (half_t)(k1 * c - k2 * s);
    K16[base + 1] = (half_t)(k1 * s + k2 * c);
}

// ---------------------------------------------------------------------------
// Tiled WMMA GEMM: C[M,N] = A[M,K](f16) @ B[K,N](f16) + bias, epilogues.
// Block 256 thr = 8 waves; tile 256x128, k-step 32.
// Wave grid 4(M) x 2(N): each wave computes 64x64 = 4x4 WMMA fragments,
// so 16 WMMAs per 16 ds_load_b128 per k-step (1:1 ratio).
// A staged row-major [m][k] (async global->LDS);
// B staged TRANSPOSED [n][k] so B fragments are contiguous 32B runs.
// ---------------------------------------------------------------------------
#define BM 256
#define BN 128
#define BK 32
#define AS_STRIDE 40     // 32 + pad; row = 80B, 16B-aligned chunks
#define BT_STRIDE 40     // 32 + pad (transposed B: 128 rows of 32 k)

template<int DO_GELU, int HAS_RESID, int HAS_OUTF, int HAS_OUTH>
__global__ void __launch_bounds__(256)
gemm_wmma_kernel(const half_t* __restrict__ A, const half_t* __restrict__ Bw,
                 const float* __restrict__ bias, const float* __restrict__ resid,
                 float* __restrict__ outF, half_t* __restrict__ outH,
                 int M, int N, int K) {
    __shared__ __align__(16) half_t As [BM * AS_STRIDE];
    __shared__ __align__(16) half_t BsT[BN * BT_STRIDE];

    const int tid   = threadIdx.x;
    const int lane  = tid & 31;
    const int wave  = tid >> 5;
    const int wm    = wave & 3;          // 4 waves along M (64 rows each)
    const int wn    = wave >> 2;         // 2 waves along N (64 cols each)
    const int m0    = blockIdx.y * BM;
    const int n0    = blockIdx.x * BN;
    const int lhalf = (lane >> 4) & 1;
    const int l15   = lane & 15;

    v8f acc[4][4];
    #pragma unroll
    for (int i = 0; i < 4; i++)
        #pragma unroll
        for (int j = 0; j < 4; j++) acc[i][j] = v8f_zero();

    for (int kk = 0; kk < K; kk += BK) {
        // ---- stage A tile (256x32), row-major, async b128 chunks ----
        #pragma unroll
        for (int i = 0; i < 4; i++) {
            int linear = (tid + i * 256) * 8;
            int r = linear / BK, c = linear % BK;
            copy_b128_g2l(A + (size_t)(m0 + r) * K + kk + c,
                          &As[r * AS_STRIDE + c]);
        }
        // ---- stage B tile (32x128) transposed into BsT[n][k] ----
        #pragma unroll
        for (int i = 0; i < 2; i++) {
            int linear = (tid + i * 256) * 8;
            int r = linear / BN, c = linear % BN;       // r = k, c = n..n+7
            v8h v = *(const v8h*)(Bw + (size_t)(kk + r) * N + n0 + c);
            #pragma unroll
            for (int j = 0; j < 8; j++)
                BsT[(c + j) * BT_STRIDE + r] = v[j];
        }
        wait_g2l();
        __syncthreads();

        // ---- A fragments: two contiguous 8-half chunks per frag ----
        v16h afrag[4];
        #pragma unroll
        for (int mf = 0; mf < 4; mf++) {
            const int mrow = wm * 64 + mf * 16 + l15;
            afrag[mf] = load_frag16_split(&As[mrow * AS_STRIDE + lhalf * 16],
                                          &As[mrow * AS_STRIDE + lhalf * 16 + 8]);
        }
        // ---- B fragments (contiguous) + 16 WMMAs ----
        #pragma unroll
        for (int nf = 0; nf < 4; nf++) {
            const int ncol = wn * 64 + nf * 16 + l15;
            v16h bfrag = load_frag16(&BsT[ncol * BT_STRIDE + lhalf * 16]);
            #pragma unroll
            for (int mf = 0; mf < 4; mf++) {
                acc[mf][nf] = __builtin_amdgcn_wmma_f32_16x16x32_f16(
                    false, afrag[mf], false, bfrag, (short)0, acc[mf][nf],
                    false, false);
            }
        }
        __syncthreads();
    }

    // ---- epilogue: compile-time specialized ----
    #pragma unroll
    for (int mf = 0; mf < 4; mf++) {
        #pragma unroll
        for (int nf = 0; nf < 4; nf++) {
            const int n = n0 + wn * 64 + nf * 16 + l15;
            const float bv = bias[n];
            #pragma unroll
            for (int r = 0; r < 8; r++) {
                const int m = m0 + wm * 64 + mf * 16 + r + lhalf * 8;
                float val = acc[mf][nf][r] + bv;
                if (DO_GELU) val = 0.5f * val * (1.0f + erff(val * 0.70710678f));
                if (HAS_RESID) val += resid[(size_t)m * N + n];
                if (HAS_OUTF) outF[(size_t)m * N + n] = val;
                if (HAS_OUTH) outH[(size_t)m * N + n] = (half_t)val;
            }
        }
    }
}

// ---------------------------------------------------------------------------
// Causal flash attention, per (b,h): 128 q-rows per block, 8 waves x 16 rows.
// K tile staged [key][dh] (async copy, fragment K-dim contiguous);
// V tile staged TRANSPOSED [dh][key] so P@V B-fragments are contiguous.
// Mask branch only near the diagonal; softmax reductions via DPP (VALU).
// ---------------------------------------------------------------------------
#define KT_STRIDE 72     // 64 + pad (row = 144B, 16B aligned chunks)
#define VT_STRIDE 40     // 32 + pad (transposed V: 64 rows of 32 keys)
#define PB_STRIDE 40     // 32 + pad (row = 80B -> 16B-aligned chunks)

__global__ void __launch_bounds__(256)
attn_wmma_kernel(const half_t* __restrict__ Q16, const half_t* __restrict__ K16,
                 const half_t* __restrict__ V16, half_t* __restrict__ O16) {
    __shared__ __align__(16) half_t Kt [32 * KT_STRIDE];
    __shared__ __align__(16) half_t VtT[64 * VT_STRIDE];
    __shared__ __align__(16) half_t Pb [8 * 16 * PB_STRIDE];

    const int tid   = threadIdx.x;
    const int lane  = tid & 31;
    const int wave  = tid >> 5;
    const int lhalf = (lane >> 4) & 1;
    const int l15   = lane & 15;
    const int b     = blockIdx.y / H;
    const int h     = blockIdx.y % H;
    const int qbase = blockIdx.x * 128;
    const int q0    = qbase + wave * 16;

    // Q fragments: 16 rows x 64 dh = two 16x32 A-frags (Q pre-scaled)
    v16h aQ[2];
    {
        const size_t rowb = ((size_t)(b * T + q0 + l15)) * D + h * DH;
        #pragma unroll
        for (int c = 0; c < 2; c++)
            aQ[c] = load_frag16_split(Q16 + rowb + c * 32 + lhalf * 16,
                                      Q16 + rowb + c * 32 + lhalf * 16 + 8);
    }

    v8f Oacc[4];
    #pragma unroll
    for (int i = 0; i < 4; i++) Oacc[i] = v8f_zero();
    float mstat[8], lstat[8];
    #pragma unroll
    for (int r = 0; r < 8; r++) { mstat[r] = -1e30f; lstat[r] = 0.0f; }

    half_t* Pw = &Pb[wave * 16 * PB_STRIDE];
    const int ktend = qbase + 128;   // causal upper bound for this block

    for (int kt = 0; kt < ktend; kt += 32) {
        // cooperative K/V tile load (32 keys x 64 dh each)
        {
            const int r = tid >> 3, c = (tid & 7) * 8;   // r = key, c = dh..dh+7
            const size_t kaddr = ((size_t)(b * T + kt + r)) * D + h * DH + c;
            copy_b128_g2l(K16 + kaddr, &Kt[r * KT_STRIDE + c]);
            v8h vv = *(const v8h*)(V16 + kaddr);
            #pragma unroll
            for (int j = 0; j < 8; j++)
                VtT[(c + j) * VT_STRIDE + r] = vv[j];     // transpose into [dh][key]
        }
        wait_g2l();
        __syncthreads();

        // S = Q @ K^T : 16x32 (two 16x16 frags), K-dim = DH = 2 chunks of 32
        v8f s[2];
        s[0] = v8f_zero(); s[1] = v8f_zero();
        #pragma unroll
        for (int nt = 0; nt < 2; nt++) {
            const int krow = nt * 16 + l15;      // key index within tile
            #pragma unroll
            for (int c = 0; c < 2; c++) {
                v16h bK = load_frag16(&Kt[krow * KT_STRIDE + c * 32 + lhalf * 16]);
                s[nt] = __builtin_amdgcn_wmma_f32_16x16x32_f16(
                    false, aQ[c], false, bK, (short)0, s[nt], false, false);
            }
        }

        // causal mask only needed for tiles touching the diagonal (wave-uniform)
        if (kt + 31 > q0) {
            #pragma unroll
            for (int r = 0; r < 8; r++) {
                const int rowg = q0 + r + lhalf * 8;
                if (kt + l15 > rowg)      s[0][r] = -1e30f;
                if (kt + 16 + l15 > rowg) s[1][r] = -1e30f;
            }
        }

        // online softmax (DPP butterfly reductions, VALU only)
        #pragma unroll
        for (int r = 0; r < 8; r++) {
            const float mloc  = red_max16(fmaxf(s[0][r], s[1][r]));
            const float mnew  = fmaxf(mstat[r], mloc);
            const float scale = __expf(mstat[r] - mnew);
            const float p0 = __expf(s[0][r] - mnew);
            const float p1 = __expf(s[1][r] - mnew);
            const float ls = red_sum16(p0 + p1);
            lstat[r] = lstat[r] * scale + ls;
            mstat[r] = mnew;
            #pragma unroll
            for (int nf = 0; nf < 4; nf++) Oacc[nf][r] *= scale;
            // P (C-layout) -> per-wave LDS staging (row-major 16x32)
            const int rowl = r + lhalf * 8;
            Pw[rowl * PB_STRIDE + l15]      = (half_t)p0;
            Pw[rowl * PB_STRIDE + 16 + l15] = (half_t)p1;
        }
        __asm__ volatile("s_wait_dscnt 0" ::: "memory");

        // reload P as 16x32 A-fragment (two contiguous 8-half chunks)
        v16h aP = load_frag16_split(&Pw[l15 * PB_STRIDE + lhalf * 16],
                                    &Pw[l15 * PB_STRIDE + lhalf * 16 + 8]);

        // O += P @ V  (16x64: 4 frags; VtT rows are contiguous key-runs)
        #pragma unroll
        for (int nf = 0; nf < 4; nf++) {
            const int ncol = nf * 16 + l15;
            v16h bV = load_frag16(&VtT[ncol * VT_STRIDE + lhalf * 16]);
            Oacc[nf] = __builtin_amdgcn_wmma_f32_16x16x32_f16(
                false, aP, false, bV, (short)0, Oacc[nf], false, false);
        }
        __syncthreads();
    }

    // normalize and store f16 attention output in [B*T, D] layout
    #pragma unroll
    for (int nf = 0; nf < 4; nf++)
        #pragma unroll
        for (int r = 0; r < 8; r++) {
            const int rowg = q0 + r + lhalf * 8;
            const size_t addr = ((size_t)(b * T + rowg)) * D + h * DH + nf * 16 + l15;
            O16[addr] = (half_t)(Oacc[nf][r] / lstat[r]);
        }
}

// ---------------------------------------------------------------------------
// Host orchestration
// ---------------------------------------------------------------------------
extern "C" void kernel_launch(void* const* d_in, const int* in_sizes, int n_in,
                              void* d_out, int out_size, void* d_ws, size_t ws_size,
                              hipStream_t stream) {
    (void)in_sizes; (void)n_in; (void)out_size; (void)ws_size;

    const float* x    = (const float*)d_in[0];
    const float* cosa = (const float*)d_in[1];
    const float* sina = (const float*)d_in[2];
    // d_in[3] = causal_mask (handled analytically in the attention kernel)
    const float* Wq = (const float*)d_in[4];
    const float* bq = (const float*)d_in[5];
    const float* Wk = (const float*)d_in[6];
    const float* bk = (const float*)d_in[7];
    const float* Wv = (const float*)d_in[8];
    const float* bv = (const float*)d_in[9];
    const float* Wo = (const float*)d_in[10];
    const float* bo = (const float*)d_in[11];
    const float* g1 = (const float*)d_in[12];
    const float* b1n= (const float*)d_in[13];
    const float* g2 = (const float*)d_in[14];
    const float* b2n= (const float*)d_in[15];
    const float* W1 = (const float*)d_in[16];
    const float* bf1= (const float*)d_in[17];
    const float* W2 = (const float*)d_in[18];
    const float* bf2= (const float*)d_in[19];
    float* out = (float*)d_out;

    char* ws = (char*)d_ws;
    const size_t MB = 1024ull * 1024ull;
    half_t* Wq16 = (half_t*)(ws +   0 * MB);
    half_t* Wk16 = (half_t*)(ws +   2 * MB);
    half_t* Wv16 = (half_t*)(ws +   4 * MB);
    half_t* Wo16 = (half_t*)(ws +   6 * MB);
    half_t* W116 = (half_t*)(ws +   8 * MB);   //  8 MB
    half_t* W216 = (half_t*)(ws +  16 * MB);   //  8 MB
    half_t* hA   = (half_t*)(ws +  24 * MB);   //  8 MB  ln1 out
    float*  Qf   = (float* )(ws +  32 * MB);   // 16 MB
    float*  Kf   = (float* )(ws +  48 * MB);   // 16 MB
    float*  Vf   = (float* )(ws +  64 * MB);   // 16 MB
    half_t* Q16  = (half_t*)(ws +  80 * MB);
    half_t* K16  = (half_t*)(ws +  88 * MB);
    half_t* V16  = (half_t*)(ws +  96 * MB);
    half_t* A16  = (half_t*)(ws + 104 * MB);   // attention out f16
    float*  xo   = (float* )(ws + 112 * MB);   // 16 MB residual-1 out
    half_t* h2   = (half_t*)(ws + 128 * MB);   // ln2 out
    half_t* g16  = (half_t*)(ws + 136 * MB);   // 32 MB ffn1 out (4096x4096)

    const int Mrows = Bc * T;                  // 4096

    // weights -> f16
    f32_to_f16_kernel<<<(D * D + 255) / 256, 256, 0, stream>>>(Wq, Wq16, D * D);
    f32_to_f16_kernel<<<(D * D + 255) / 256, 256, 0, stream>>>(Wk, Wk16, D * D);
    f32_to_f16_kernel<<<(D * D + 255) / 256, 256, 0, stream>>>(Wv, Wv16, D * D);
    f32_to_f16_kernel<<<(D * D + 255) / 256, 256, 0, stream>>>(Wo, Wo16, D * D);
    f32_to_f16_kernel<<<(D * 4 * D + 255) / 256, 256, 0, stream>>>(W1, W116, D * 4 * D);
    f32_to_f16_kernel<<<(D * 4 * D + 255) / 256, 256, 0, stream>>>(W2, W216, D * 4 * D);

    // LN1
    layernorm_kernel<<<Mrows, 256, 0, stream>>>(x, g1, b1n, hA);

    // QKV projections (WMMA): <gelu, resid, outF, outH>
    dim3 gq(D / BN, Mrows / BM);
    gemm_wmma_kernel<0,0,1,0><<<gq, 256, 0, stream>>>(hA, Wq16, bq, nullptr, Qf, nullptr, Mrows, D, D);
    gemm_wmma_kernel<0,0,1,0><<<gq, 256, 0, stream>>>(hA, Wk16, bk, nullptr, Kf, nullptr, Mrows, D, D);
    gemm_wmma_kernel<0,0,1,0><<<gq, 256, 0, stream>>>(hA, Wv16, bv, nullptr, Vf, nullptr, Mrows, D, D);

    // RoPE + f16 conversion (Q pre-scaled by 1/sqrt(DH))
    const int ropeN = Bc * T * H * NB;
    rope_qk_kernel<<<(ropeN + 255) / 256, 256, 0, stream>>>(Qf, Kf, cosa, sina, Q16, K16, ropeN);
    f32_to_f16_kernel<<<(Mrows * D + 255) / 256, 256, 0, stream>>>(Vf, V16, Mrows * D);

    // causal flash attention (WMMA)
    dim3 ga(T / 128, Bc * H);
    attn_wmma_kernel<<<ga, 256, 0, stream>>>(Q16, K16, V16, A16);

    // output projection + residual
    gemm_wmma_kernel<0,1,1,0><<<gq, 256, 0, stream>>>(A16, Wo16, bo, x, xo, nullptr, Mrows, D, D);

    // LN2
    layernorm_kernel<<<Mrows, 256, 0, stream>>>(xo, g2, b2n, h2);

    // FFN1 (+exact GELU), f16 out
    dim3 gf1(4 * D / BN, Mrows / BM);
    gemm_wmma_kernel<1,0,0,1><<<gf1, 256, 0, stream>>>(h2, W116, bf1, nullptr, nullptr, g16, Mrows, 4 * D, D);

    // FFN2 + residual -> final fp32 output
    gemm_wmma_kernel<0,1,1,0><<<gq, 256, 0, stream>>>(g16, W216, bf2, xo, out, nullptr, Mrows, D, 4 * D);
}
